// ConvGraphQNN_65481071397797
// MI455X (gfx1250) — matrix operand order repaired and checked by reading, not compiled, not run on hardware
//
#include <hip/hip_runtime.h>
#include <math.h>

typedef __attribute__((ext_vector_type(2)))  float    v2f;
typedef __attribute__((ext_vector_type(8)))  float    v8f;
typedef __attribute__((ext_vector_type(16))) _Float16 v16h;

#define NSAMP 8192
#define KTOT  64          // 8*8 kernel flattened
#define EPSV  1e-12f

// ---------------------------------------------------------------------------
// Phase A: logits = data[n,:] . w  + b   via WMMA, acts = sigmoid(logits)
// One wave computes a 16-sample tile: D(16x16) = A(16x4) * B(4x16) accumulated
// over K=64 in steps of 4 (full-precision f32 WMMA). B has w replicated into
// every column, so every column of D equals the logit vector.
// ---------------------------------------------------------------------------
__global__ __launch_bounds__(256)
void conv_acts_wmma_kernel(const float* __restrict__ data,
                           const float* __restrict__ w,
                           const float* __restrict__ bias_p,
                           float* __restrict__ acts_out,
                           float* __restrict__ an_out) {
    const int lane = threadIdx.x & 31;
    const int waveInBlock = threadIdx.x >> 5;
    const int tile = blockIdx.x * 8 + waveInBlock;     // 512 tiles of 16 rows
    const int row  = tile * 16 + (lane & 15);          // sample index for A
    const bool hi  = (lane >= 16);

    v8f c = {};

#if __has_builtin(__builtin_amdgcn_wmma_f32_16x16x4_f32)
    // 32-bit A 16x4 layout: lanes 0-15 -> {V0:K0, V1:K1}, lanes 16-31 -> {V0:K2, V1:K3}
    // => per lane a contiguous float2 at k + (hi?2:0). B rows replicated across columns.
    const int koff = hi ? 2 : 0;
    for (int k = 0; k < KTOT; k += 4) {
        v2f a  = *(const v2f*)(data + (size_t)row * KTOT + k + koff);
        v2f bb = *(const v2f*)(w + k + koff);
        c = __builtin_amdgcn_wmma_f32_16x16x4_f32(false, a, false, bb,
                                                  (short)0, c, false, false);
    }
#else
    // Fallback (codegen-confirmed builtin): f16 16x16x32, two K-chunks of 32.
    for (int kb = 0; kb < KTOT; kb += 32) {
        v16h A, B;
        #pragma unroll
        for (int h = 0; h < 16; ++h) {
            // 16-bit A 16x32 layout -> K index for half-element h of this lane
            int ka = ((h >> 3) * 16) + (hi ? 8 : 0) + (((h >> 1) & 3) * 2) + (h & 1);
            A[h] = (_Float16)data[(size_t)row * KTOT + kb + ka];
            // B 32x16: same K order as A (element h -> K = (h>>3)*16 + (hi?8:0) + (h&7));
            // columns identical, value = w[k]
            int kbrow = ((h >> 3) * 16) + (hi ? 8 : 0) + (h & 7);
            B[h] = (_Float16)w[kb + kbrow];
        }
        c = __builtin_amdgcn_wmma_f32_16x16x32_f16(false, A, false, B,
                                                   (short)0, c, false, false);
    }
#endif

    // D layout (f32 16x16): lane n (col), VGPR r -> M = r + (hi ? 8 : 0).
    // All columns equal; let column-0 lanes (lane 0 and lane 16) publish.
    const float bias = bias_p[0];
    if ((lane & 15) == 0) {
        const int mbase = tile * 16 + (hi ? 8 : 0);
        #pragma unroll
        for (int r = 0; r < 8; ++r) {
            float logit = c[r] + bias;                 // CONV_THRESHOLD == 0
            float act   = 1.0f / (1.0f + __expf(-logit));
            acts_out[mbase + r] = act;
            an_out[mbase + r]   = act / (fabsf(act) + EPSV);
        }
    }
}

// ---------------------------------------------------------------------------
// Phase B: fid_ij = (an_i*an_j)^2 ; edge iff (fid>=0.8 || fid>=0.5) && i!=j
// out[i] = deg>0 ? sum_j(acts_j)/max(deg,1) : acts_i
// LDS-tiled sweep over j; 8192 inner iters per thread, fully L2/LDS resident.
// ---------------------------------------------------------------------------
__global__ __launch_bounds__(256)
void graph_neighbor_mean_kernel(const float* __restrict__ acts,
                                const float* __restrict__ an,
                                float* __restrict__ out) {
    __shared__ float sh_a [256];
    __shared__ float sh_an[256];

    const int i  = blockIdx.x * 256 + threadIdx.x;
    const float ai  = acts[i];
    const float ani = an[i];

    float deg = 0.0f;
    float sum = 0.0f;

    for (int t = 0; t < NSAMP; t += 256) {
        __syncthreads();
        sh_a [threadIdx.x] = acts[t + threadIdx.x];
        sh_an[threadIdx.x] = an  [t + threadIdx.x];
        __syncthreads();

        #pragma unroll 8
        for (int jj = 0; jj < 256; ++jj) {
            const int   j   = t + jj;
            const float aj  = sh_a[jj];
            const float p   = ani * sh_an[jj];
            const float fid = p * p;
            const bool edge = ((fid >= 0.8f) || (fid >= 0.5f)) && (j != i);
            deg += edge ? 1.0f : 0.0f;
            sum += edge ? aj   : 0.0f;
        }
    }

    out[i] = (deg > 0.0f) ? (sum / fmaxf(deg, 1.0f)) : ai;
}

// ---------------------------------------------------------------------------
extern "C" void kernel_launch(void* const* d_in, const int* in_sizes, int n_in,
                              void* d_out, int out_size, void* d_ws, size_t ws_size,
                              hipStream_t stream) {
    const float* data = (const float*)d_in[0];   // [8192,8,8]
    const float* w    = (const float*)d_in[1];   // [1,1,8,8]
    const float* b    = (const float*)d_in[2];   // [1]
    float*       out  = (float*)d_out;           // [8192]

    float* acts = (float*)d_ws;                  // 8192 f32
    float* an   = acts + NSAMP;                  // 8192 f32

    // 512 tiles of 16 samples; 8 waves (256 thr) per block -> 64 blocks
    conv_acts_wmma_kernel<<<NSAMP / 16 / 8, 256, 0, stream>>>(data, w, b, acts, an);
    // 8192 threads, one output element each
    graph_neighbor_mean_kernel<<<NSAMP / 256, 256, 0, stream>>>(acts, an, out);
}
